// FSGPTMoESelfAttention_57818849739125
// MI455X (gfx1250) — matrix operand order repaired and checked by reading, not compiled, not run on hardware
//
#include <hip/hip_runtime.h>
#include <hip/hip_bf16.h>

typedef __attribute__((ext_vector_type(16))) _Float16 v16h;
typedef __attribute__((ext_vector_type(8)))  _Float16 v8h;
typedef __attribute__((ext_vector_type(8)))  float    v8f;

#define BATCH   2
#define SEQ     2048
#define D_MODEL 2048
#define NHEAD   16
#define HDIM    128

// --------------------------------------------------------------------------
// CDNA5 async global->LDS copy path (GLOBAL_LOAD_ASYNC_TO_LDS_B128, ASYNCcnt)
// with sync fallback if the builtin is absent on this toolchain.
// Probe result: builtin exists; params are (AS1 v4i*, AS3 v4i*, Ii, Ii).
// --------------------------------------------------------------------------
#if defined(__has_builtin)
 #if __has_builtin(__builtin_amdgcn_global_load_async_to_lds_b128)
  #define USE_ASYNC 1
 #endif
#endif
#ifndef USE_ASYNC
#define USE_ASYNC 0
#endif

#if USE_ASYNC
typedef int v4i __attribute__((vector_size(16)));
typedef __attribute__((address_space(1))) v4i g_v4i;
typedef __attribute__((address_space(3))) v4i l_v4i;
#endif

static __device__ __forceinline__
void copy16(const _Float16* g, _Float16* l) {
#if USE_ASYNC
  __builtin_amdgcn_global_load_async_to_lds_b128(
      (g_v4i*)const_cast<_Float16*>(g), (l_v4i*)l, 0, 0);
#else
  *(v8h*)l = *(const v8h*)g;
#endif
}

static __device__ __forceinline__ void async_wait() {
#if USE_ASYNC
 #if __has_builtin(__builtin_amdgcn_s_wait_asynccnt)
  __builtin_amdgcn_s_wait_asynccnt(0);
 #else
  asm volatile("s_wait_asynccnt 0x0" ::: "memory");
 #endif
#endif
}

// ---------------------------------------------------------------------------
// Fragment loaders per CDNA5 WMMA VGPR layouts (CDNA5 ISA §7.12.2)
// A (16x32 f16): lane = row (lane&15); lane-half splits K in chunks of 8.
// B (32x16 f16): lane = col (lane&15); lane-half splits K in chunks of 8.
// C/D (16x16 f32): lane = col (lane&15); vgpr j holds row m = j + 8*(lane>>4)
// ---------------------------------------------------------------------------
static __device__ __forceinline__
v16h fragA(const _Float16* base, int stride, int row0, int k0, int lr, int lh) {
  const _Float16* p = base + (size_t)(row0 + lr) * stride + k0 + lh * 8;
  v8h lo = *(const v8h*)(p);
  v8h hi = *(const v8h*)(p + 16);
  return __builtin_shufflevector(lo, hi, 0,1,2,3,4,5,6,7,8,9,10,11,12,13,14,15);
}

static __device__ __forceinline__
v16h fragB(const _Float16* base, int stride, int col0, int k0, int lr, int lh) {
  const _Float16* p = base + (size_t)(col0 + lr) * stride + k0 + lh * 8;
  v8h lo = *(const v8h*)(p);
  v8h hi = *(const v8h*)(p + 16);
  return __builtin_shufflevector(lo, hi, 0,1,2,3,4,5,6,7,8,9,10,11,12,13,14,15);
}

static __device__ __forceinline__
v8f wmma16(v16h a, v16h b, v8f c) {
  return __builtin_amdgcn_wmma_f32_16x16x32_f16(
      false, a, false, b, (short)0, c, false, false);
}

// ---------------------------------------------------------------------------
// f32 -> f16 conversion (8 elements / thread)
// ---------------------------------------------------------------------------
__global__ __launch_bounds__(256)
void cvt_f32_f16(const float* __restrict__ in, _Float16* __restrict__ out, int n) {
  int i = (blockIdx.x * 256 + threadIdx.x) * 8;
  if (i >= n) return;
  float4 a = *(const float4*)(in + i);
  float4 b = *(const float4*)(in + i + 4);
  v8h o;
  o[0] = (_Float16)a.x; o[1] = (_Float16)a.y; o[2] = (_Float16)a.z; o[3] = (_Float16)a.w;
  o[4] = (_Float16)b.x; o[5] = (_Float16)b.y; o[6] = (_Float16)b.z; o[7] = (_Float16)b.w;
  *(v8h*)(out + i) = o;
}

// ---------------------------------------------------------------------------
// GEMM: out[M,N] = (A[M,K] @ W[N,K]^T + bias[N]) * scale
// 128x128 block tile, 8 waves (2 in M x 4 in N), wave tile 64x32.
// Double-buffered LDS, async global->LDS staging, one barrier per K-step.
// ---------------------------------------------------------------------------
template<bool OUT_F32>
__global__ __launch_bounds__(256)
void gemm_wt_bias(const _Float16* __restrict__ A, const _Float16* __restrict__ W,
                  const float* __restrict__ bias, void* __restrict__ out,
                  int M, int N, int K, float scale) {
  __shared__ __align__(16) _Float16 lA[2][128][40];   // 32 K + 8 pad (80B rows)
  __shared__ __align__(16) _Float16 lB[2][128][40];

  const int t    = threadIdx.x;
  const int wid  = t >> 5;
  const int lane = t & 31;
  const int lr   = lane & 15;
  const int lh   = lane >> 4;
  const int wm   = wid & 1;          // 0..1  (64 rows each)
  const int wn   = wid >> 1;         // 0..3  (32 cols each)
  const int bm   = blockIdx.x * 128;
  const int bn   = blockIdx.y * 128;

  v8f acc[4][2];
  #pragma unroll
  for (int mt = 0; mt < 4; ++mt)
    #pragma unroll
    for (int nt = 0; nt < 2; ++nt)
      #pragma unroll
      for (int j = 0; j < 8; ++j) acc[mt][nt][j] = 0.0f;

  auto stage = [&](int buf, int k0) {
    #pragma unroll
    for (int i = 0; i < 2; ++i) {
      int lin = t + i * 256;         // 0..511
      int r = lin >> 2;              // 0..127
      int c = (lin & 3) * 8;         // 0,8,16,24
      copy16(&A[(size_t)(bm + r) * K + k0 + c], &lA[buf][r][c]);
      copy16(&W[(size_t)(bn + r) * K + k0 + c], &lB[buf][r][c]);
    }
  };

  stage(0, 0);
  const int nk = K >> 5;
  for (int kk = 0; kk < nk; ++kk) {
    const int buf = kk & 1;
    async_wait();                    // this wave's staged tile is in LDS
    __syncthreads();                 // all waves' tiles are in LDS
    if (kk + 1 < nk) stage(buf ^ 1, (kk + 1) << 5);  // prefetch next tile

    v16h af[4], bf[2];
    #pragma unroll
    for (int mt = 0; mt < 4; ++mt)
      af[mt] = fragA(&lA[buf][0][0], 40, wm * 64 + mt * 16, 0, lr, lh);
    #pragma unroll
    for (int nt = 0; nt < 2; ++nt)
      bf[nt] = fragB(&lB[buf][0][0], 40, wn * 32 + nt * 16, 0, lr, lh);
    #pragma unroll
    for (int mt = 0; mt < 4; ++mt)
      #pragma unroll
      for (int nt = 0; nt < 2; ++nt)
        acc[mt][nt] = wmma16(af[mt], bf[nt], acc[mt][nt]);
  }

  #pragma unroll
  for (int mt = 0; mt < 4; ++mt) {
    #pragma unroll
    for (int nt = 0; nt < 2; ++nt) {
      int r0 = bm + wm * 64 + mt * 16 + lh * 8;
      int c  = bn + wn * 32 + nt * 16 + lr;
      float bb = bias[c];
      #pragma unroll
      for (int j = 0; j < 8; ++j) {
        float val = (acc[mt][nt][j] + bb) * scale;
        if (OUT_F32) ((float*)out)[(size_t)(r0 + j) * N + c] = val;
        else ((_Float16*)out)[(size_t)(r0 + j) * N + c] = (_Float16)val;
      }
    }
  }
}

// ---------------------------------------------------------------------------
// Flash attention forward, causal. 4 waves / block, 64 queries / block,
// 32-key steps. Q/K/V/ctx are f16 [B*S, D]; head h lives in cols [h*128, +128).
// K tile staged via async copies; V staged transposed (register path);
// causal mask applied only on wave-uniform diagonal tiles.
// ---------------------------------------------------------------------------
__global__ __launch_bounds__(128)
void flash_attn(const _Float16* __restrict__ Q, const _Float16* __restrict__ Km,
                const _Float16* __restrict__ V, _Float16* __restrict__ ctx) {
  __shared__ __align__(16) _Float16 lK[32][136];       // [key][d], 272B rows
  __shared__ __align__(16) _Float16 lVt[128][40];      // [d][key] (transposed)
  __shared__ __align__(16) _Float16 lP[4][16][40];     // per-wave P (16x32)

  const int t    = threadIdx.x;
  const int wid  = t >> 5;
  const int lane = t & 31;
  const int lr   = lane & 15;
  const int lh   = lane >> 4;
  const int b    = blockIdx.z;
  const int h    = blockIdx.y;
  const int q0   = blockIdx.x * 64;

  const size_t rowbase = (size_t)b * SEQ;
  const _Float16* Qg = Q  + rowbase * D_MODEL + h * HDIM;
  const _Float16* Kg = Km + rowbase * D_MODEL + h * HDIM;
  const _Float16* Vg = V  + rowbase * D_MODEL + h * HDIM;

  // Preload this wave's Q tile (16 rows x 128 d) as 4 A-fragments (registers).
  v16h qf[4];
  #pragma unroll
  for (int dd = 0; dd < 4; ++dd)
    qf[dd] = fragA(Qg, D_MODEL, q0 + wid * 16, dd * 32, lr, lh);

  float run_m[8], run_l[8];
  v8f o[8];
  #pragma unroll
  for (int j = 0; j < 8; ++j) { run_m[j] = -3.0e38f; run_l[j] = 0.0f; }
  #pragma unroll
  for (int ct = 0; ct < 8; ++ct)
    #pragma unroll
    for (int j = 0; j < 8; ++j) o[ct][j] = 0.0f;

  const int qw   = q0 + wid * 16;               // first query row of this wave
  const int qmax = qw + 15;                     // last query row of this wave
  const int kend = q0 + 64;                     // causal bound for the block

  for (int kt = 0; kt < kend; kt += 32) {
    // ---- stage K (async, row-major) and V (transposed) into LDS ----
    #pragma unroll
    for (int i = 0; i < 4; ++i) {
      int lin = t + i * 128;                    // 0..511
      int key = lin >> 4;                       // 0..31
      int c0  = (lin & 15) * 8;                 // 0..120
      copy16(&Kg[(size_t)(kt + key) * D_MODEL + c0], &lK[key][c0]);
      v8h vv = *(const v8h*)&Vg[(size_t)(kt + key) * D_MODEL + c0];
      #pragma unroll
      for (int e = 0; e < 8; ++e) lVt[c0 + e][key] = vv[e];
    }
    async_wait();
    __syncthreads();

    if (kt <= qmax) {                           // wave-uniform predicate
      // ---- scores S = Q @ K^T  (16 x 32) ----
      v8f s[2];
      #pragma unroll
      for (int ns = 0; ns < 2; ++ns) {
        #pragma unroll
        for (int j = 0; j < 8; ++j) s[ns][j] = 0.0f;
        #pragma unroll
        for (int dd = 0; dd < 4; ++dd) {
          v16h bk = fragB(&lK[0][0], 136, ns * 16, dd * 32, lr, lh);
          s[ns] = wmma16(qf[dd], bk, s[ns]);
        }
      }
      // ---- causal mask: only on diagonal tiles (wave-uniform branch) ----
      if (kt + 31 > qw) {
        #pragma unroll
        for (int ns = 0; ns < 2; ++ns)
          #pragma unroll
          for (int j = 0; j < 8; ++j) {
            int key = kt + ns * 16 + lr;
            int qq  = qw + lh * 8 + j;
            if (key > qq) s[ns][j] = -3.0e38f;
          }
      }
      // ---- online softmax (row reductions over 16-lane halves) ----
      float alpha[8];
      #pragma unroll
      for (int j = 0; j < 8; ++j) {
        float mx = fmaxf(s[0][j], s[1][j]);
        #pragma unroll
        for (int d = 1; d < 16; d <<= 1) mx = fmaxf(mx, __shfl_xor(mx, d, 32));
        float mn = fmaxf(run_m[j], mx);
        alpha[j] = __expf(run_m[j] - mn);
        run_m[j] = mn;
        float p0 = __expf(s[0][j] - mn);
        float p1 = __expf(s[1][j] - mn);
        float rs = p0 + p1;
        #pragma unroll
        for (int d = 1; d < 16; d <<= 1) rs += __shfl_xor(rs, d, 32);
        run_l[j] = run_l[j] * alpha[j] + rs;
        // C-layout -> LDS so P can be re-read in A-layout
        lP[wid][lh * 8 + j][lr]      = (_Float16)p0;
        lP[wid][lh * 8 + j][16 + lr] = (_Float16)p1;
      }
      #pragma unroll
      for (int ct = 0; ct < 8; ++ct)
        #pragma unroll
        for (int j = 0; j < 8; ++j) o[ct][j] *= alpha[j];

      asm volatile("" ::: "memory");   // keep LDS P stores before the reload
      // ---- O += P @ V  (P: 16x32 A-frag, V^T columns contiguous in lVt) ----
      v16h pf = fragA(&lP[wid][0][0], 40, 0, 0, lr, lh);
      #pragma unroll
      for (int ct = 0; ct < 8; ++ct) {
        v16h bv = fragB(&lVt[0][0], 40, ct * 16, 0, lr, lh);
        o[ct] = wmma16(pf, bv, o[ct]);
      }
    }
    __syncthreads();
  }

  // ---- normalize and store context (f16) ----
  #pragma unroll
  for (int ct = 0; ct < 8; ++ct) {
    #pragma unroll
    for (int j = 0; j < 8; ++j) {
      float val = o[ct][j] / run_l[j];
      size_t row = rowbase + q0 + wid * 16 + lh * 8 + j;
      ctx[row * D_MODEL + h * HDIM + ct * 16 + lr] = (_Float16)val;
    }
  }
}

// ---------------------------------------------------------------------------
extern "C" void kernel_launch(void* const* d_in, const int* in_sizes, int n_in,
                              void* d_out, int out_size, void* d_ws, size_t ws_size,
                              hipStream_t stream) {
  const float* x  = (const float*)d_in[0];
  const float* Wq = (const float*)d_in[1];
  const float* bq = (const float*)d_in[2];
  const float* Wk = (const float*)d_in[3];
  const float* bk = (const float*)d_in[4];
  const float* Wv = (const float*)d_in[5];
  const float* bv = (const float*)d_in[6];
  const float* Wo = (const float*)d_in[7];
  const float* bo = (const float*)d_in[8];

  const size_t nx = (size_t)BATCH * SEQ * D_MODEL;   // 8388608
  const size_t nw = (size_t)D_MODEL * D_MODEL;       // 4194304

  _Float16* ws  = (_Float16*)d_ws;
  _Float16* xh  = ws;
  _Float16* wqh = xh  + nx;
  _Float16* wkh = wqh + nw;
  _Float16* wvh = wkh + nw;
  _Float16* woh = wvh + nw;
  _Float16* qh  = woh + nw;
  _Float16* kh  = qh  + nx;
  _Float16* vh  = kh  + nx;
  _Float16* ch  = vh  + nx;

  cvt_f32_f16<<<(int)(nx / 2048), 256, 0, stream>>>(x,  xh,  (int)nx);
  cvt_f32_f16<<<(int)(nw / 2048), 256, 0, stream>>>(Wq, wqh, (int)nw);
  cvt_f32_f16<<<(int)(nw / 2048), 256, 0, stream>>>(Wk, wkh, (int)nw);
  cvt_f32_f16<<<(int)(nw / 2048), 256, 0, stream>>>(Wv, wvh, (int)nw);
  cvt_f32_f16<<<(int)(nw / 2048), 256, 0, stream>>>(Wo, woh, (int)nw);

  dim3 gg(BATCH * SEQ / 128, D_MODEL / 128);
  const float qscale = 0.08838834764831845f;   // 1/sqrt(128)
  gemm_wt_bias<false><<<gg, 256, 0, stream>>>(xh, wqh, bq, qh,
      BATCH * SEQ, D_MODEL, D_MODEL, qscale);
  gemm_wt_bias<false><<<gg, 256, 0, stream>>>(xh, wkh, bk, kh,
      BATCH * SEQ, D_MODEL, D_MODEL, 1.0f);
  gemm_wt_bias<false><<<gg, 256, 0, stream>>>(xh, wvh, bv, vh,
      BATCH * SEQ, D_MODEL, D_MODEL, 1.0f);

  dim3 fg(SEQ / 64, NHEAD, BATCH);
  flash_attn<<<fg, 128, 0, stream>>>(qh, kh, vh, ch);

  gemm_wt_bias<true><<<gg, 256, 0, stream>>>(ch, woh, bo, d_out,
      BATCH * SEQ, D_MODEL, D_MODEL, 1.0f);
}
